// MoNetUnet_68805376082187
// MI455X (gfx1250) — matrix operand order repaired and checked by reading, not compiled, not run on hardware
//
#include <hip/hip_runtime.h>
#include <hip/hip_bf16.h>
#include <math.h>

// ---------------------------------------------------------------------------
// MoNet U-Net (GMMConv) for gfx1250.
// Dense GEMMs (x @ [g|root]) run on v_wmma_f32_16x16x32_bf16 (f32 accumulate).
// Edge message passing is gather + global f32 atomics (memory-bound path).
// ---------------------------------------------------------------------------

typedef __attribute__((ext_vector_type(16))) __bf16 v16bf;
typedef __attribute__((ext_vector_type(8)))  __bf16 v8bf;
typedef __attribute__((ext_vector_type(8)))  float  v8f;

#define CN7 163842
#define CN6 40962
#define CN5 10242
#define CE7 (6 * CN7)
#define CE6 (6 * CN6)
#define CE5 (6 * CN5)

// ---------------------------------------------------------------------------
// Kernels
// ---------------------------------------------------------------------------

__global__ void zero_f32(float* __restrict__ p, size_t n) {
    size_t i = (size_t)blockIdx.x * blockDim.x + threadIdx.x;
    if (i < n) p[i] = 0.0f;
}

__global__ void f32_to_bf16(const float* __restrict__ x, __bf16* __restrict__ y, size_t n) {
    size_t i = (size_t)blockIdx.x * blockDim.x + threadIdx.x;
    if (i < n) y[i] = (__bf16)x[i];
}

// Build combined transposed weight Wt[col][ci], col in [0,4M):
//   col <  3M : g[ci, col]       (g is [cin, 3M] row-major)
//   col >= 3M : root[ci, col-3M] (root is [cin, M] row-major)
__global__ void prep_weights(const float* __restrict__ g, const float* __restrict__ root,
                             __bf16* __restrict__ Wt, int cin, int M) {
    int t = blockIdx.x * blockDim.x + threadIdx.x;
    int total = 4 * M * cin;
    if (t >= total) return;
    int col = t / cin;
    int ci  = t - col * cin;
    float v = (col < 3 * M) ? g[(size_t)ci * (3 * M) + col]
                            : root[(size_t)ci * M + (col - 3 * M)];
    Wt[t] = (__bf16)v;
}

// Per-lane A-fragment loader (16-bit A 16x32 layout):
//   lanes 0-15 hold K = 0..7 and 16..23; lanes 16-31 hold K = 8..15 and 24..31.
__device__ __forceinline__ v16bf load_a_frag(const __bf16* px) {
    v8bf a0 = *reinterpret_cast<const v8bf*>(px);        // K = alo..alo+7
    v8bf a1 = *reinterpret_cast<const v8bf*>(px + 16);   // K = 16+alo..16+alo+7
    return __builtin_shufflevector(a0, a1, 0, 1, 2, 3, 4, 5, 6, 7,
                                   8, 9, 10, 11, 12, 13, 14, 15);
}

// Y[N, cols] = X[N, cin](bf16) @ W[cin, cols](bf16), f32 accumulate.
// One wave per 16x16 output tile; K-loop stride 32 using wmma_f32_16x16x32_bf16.
// Wt is column-major: Wt[col*cin + k]. Full row-tiles take a guard-free fast
// path (wave-uniform branch) so the hot loop is pure load + wmma.
__global__ void gemm_xw_bf16(const __bf16* __restrict__ X, const __bf16* __restrict__ Wt,
                             float* __restrict__ Y, int N, int cin, int cols, int tiles) {
    int wave = blockIdx.x * (blockDim.x >> 5) + (threadIdx.x >> 5);
    if (wave >= tiles) return;                  // wave-uniform exit, EXEC stays full
    int lane   = threadIdx.x & 31;
    int tilesC = cols >> 4;
    int tm = wave / tilesC, tc = wave - tm * tilesC;
    int r0 = tm << 4, c0 = tc << 4;
    int rl   = lane & 15;
    int half = lane >> 4;                       // 0: lanes 0-15, 1: lanes 16-31
    int r = r0 + rl;                            // A row for this lane
    int c = c0 + rl;                            // B/D column for this lane
    int alo  = half * 8;                        // A: K chunk base (0 or 8)
    int bofs = half * 8;                        // B: K chunk base (0 or 8)

    const __bf16* pa = X + (size_t)r * cin + alo;
    const __bf16* pb = Wt + (size_t)c * cin + bofs;

    v8f acc = {0.f, 0.f, 0.f, 0.f, 0.f, 0.f, 0.f, 0.f};
    int rbase = r0 + half * 8;

    if (r0 + 16 <= N) {
        // ---- fast path: full tile, no guards anywhere ----
        for (int k0 = 0; k0 < cin; k0 += 32) {
            v16bf a = load_a_frag(pa + k0);
            v16bf b = load_a_frag(pb + k0);
            acc = __builtin_amdgcn_wmma_f32_16x16x32_bf16(false, a, false, b,
                                                          (short)0, acc, false, false);
        }
#pragma unroll
        for (int i = 0; i < 8; ++i)
            Y[(size_t)(rbase + i) * cols + c] = acc[i];
    } else {
        // ---- tail path: partial row tile (rare) ----
        for (int k0 = 0; k0 < cin; k0 += 32) {
            v16bf a;
            if (r < N) {
                a = load_a_frag(pa + k0);
            } else {
#pragma unroll
                for (int j = 0; j < 16; ++j) a[j] = (__bf16)0.0f;
            }
            v16bf b = load_a_frag(pb + k0);
            acc = __builtin_amdgcn_wmma_f32_16x16x32_bf16(false, a, false, b,
                                                          (short)0, acc, false, false);
        }
#pragma unroll
        for (int i = 0; i < 8; ++i) {
            int row = rbase + i;
            if (row < N) Y[(size_t)row * cols + c] = acc[i];
        }
    }
}

// w[e,k] = exp(-0.5 * sum_d (ea[e,d]-mu[k,d])^2 / (eps + sigma[k,d]^2))
__global__ void edge_weights(const float* __restrict__ ea, const float* __restrict__ mu,
                             const float* __restrict__ sigma, float* __restrict__ w, int E) {
    int e = blockIdx.x * blockDim.x + threadIdx.x;
    if (e >= E) return;
    float a0 = ea[2 * e], a1 = ea[2 * e + 1];
#pragma unroll
    for (int k = 0; k < 3; ++k) {
        float d0 = a0 - mu[2 * k], d1 = a1 - mu[2 * k + 1];
        float s0 = sigma[2 * k], s1 = sigma[2 * k + 1];
        float t = -0.5f * d0 * d0 / (1e-15f + s0 * s0)
                  - 0.5f * d1 * d1 / (1e-15f + s1 * s1);
        w[3 * e + k] = __expf(t);
    }
}

// agg[dst, m] += sum_k w[e,k] * Y[src, k*M + m]; one thread per (edge, m)
__global__ void scatter_msgs(const int* __restrict__ edges, const float* __restrict__ Y,
                             const float* __restrict__ w, float* __restrict__ agg,
                             int E, int M, int cols) {
    int t = blockIdx.x * blockDim.x + threadIdx.x;
    if (t >= E * M) return;
    int e = t / M, m = t - e * M;
    int s = edges[e];          // edges[0, e]
    int d = edges[E + e];      // edges[1, e]
    const float* ys = Y + (size_t)s * cols;
    float v = w[3 * e] * ys[m] + w[3 * e + 1] * ys[M + m] + w[3 * e + 2] * ys[2 * M + m];
    atomicAdd(agg + (size_t)d * M + m, v);
}

__global__ void count_edges(const int* __restrict__ edges, float* __restrict__ cnt, int E) {
    int e = blockIdx.x * blockDim.x + threadIdx.x;
    if (e < E) atomicAdd(&cnt[edges[E + e]], 1.0f);
}

// x_out[n,m] = relu( agg[n,m]/max(cnt[n],1) + Y[n, 3M+m] + bias[m] )
__global__ void finalize_k(const float* __restrict__ agg, const float* __restrict__ cnt,
                           const float* __restrict__ Y, const float* __restrict__ bias,
                           float* __restrict__ xout, int N, int M, int cols) {
    int t = blockIdx.x * blockDim.x + threadIdx.x;
    if (t >= N * M) return;
    int n = t / M, m = t - n * M;
    float v = agg[t] / fmaxf(cnt[n], 1.0f) + Y[(size_t)n * cols + 3 * M + m] + bias[m];
    xout[t] = fmaxf(v, 0.0f);
}

// HexPool: xbf[n,c] = bf16( max_{j<7} x[neigh[n,j], c] )
__global__ void hex_pool(const float* __restrict__ x, const int* __restrict__ neigh,
                         __bf16* __restrict__ xbf, int Nout, int C) {
    int t = blockIdx.x * blockDim.x + threadIdx.x;
    if (t >= Nout * C) return;
    int n = t / C, c = t - n * C;
    float v = x[(size_t)neigh[n * 7] * C + c];
#pragma unroll
    for (int j = 1; j < 7; ++j)
        v = fmaxf(v, x[(size_t)neigh[n * 7 + j] * C + c]);
    xbf[(size_t)n * C + c] = (__bf16)v;
}

// HexUnpool into concat buffer columns [0,Cx) of a Ccat-wide bf16 matrix.
__global__ void hex_unpool(const float* __restrict__ x, const int* __restrict__ ups,
                           __bf16* __restrict__ xbf, int Nsmall, int Nbig, int Cx, int Ccat) {
    int t = blockIdx.x * blockDim.x + threadIdx.x;
    if (t >= Nbig * Cx) return;
    int n = t / Cx, c = t - n * Cx;
    float v;
    if (n < Nsmall) {
        v = x[(size_t)n * Cx + c];
    } else {
        int i = n - Nsmall;
        int a = ups[2 * i], b = ups[2 * i + 1];
        v = 0.5f * (x[(size_t)a * Cx + c] + x[(size_t)b * Cx + c]);
    }
    xbf[(size_t)n * Ccat + c] = (__bf16)v;
}

// Copy skip (f32, Cs wide) into columns [off, off+Cs) of concat bf16 buffer.
__global__ void concat_skip(const float* __restrict__ skip, __bf16* __restrict__ xbf,
                            int N, int Cs, int Ccat, int off) {
    int t = blockIdx.x * blockDim.x + threadIdx.x;
    if (t >= N * Cs) return;
    int n = t / Cs, c = t - n * Cs;
    xbf[(size_t)n * Ccat + off + c] = (__bf16)skip[t];
}

// out[n,:] = log_softmax(x[n,:] @ w + b), w is [32,2] row-major
__global__ void fc_logsoftmax(const float* __restrict__ x, const float* __restrict__ w,
                              const float* __restrict__ b, float* __restrict__ out, int N) {
    int n = blockIdx.x * blockDim.x + threadIdx.x;
    if (n >= N) return;
    float z0 = b[0], z1 = b[1];
#pragma unroll
    for (int c = 0; c < 32; ++c) {
        float xc = x[(size_t)n * 32 + c];
        z0 += xc * w[2 * c];
        z1 += xc * w[2 * c + 1];
    }
    float m  = fmaxf(z0, z1);
    float ls = m + __logf(__expf(z0 - m) + __expf(z1 - m));
    out[2 * n]     = z0 - ls;
    out[2 * n + 1] = z1 - ls;
}

// ---------------------------------------------------------------------------
// Host orchestration
// ---------------------------------------------------------------------------

struct GmmP { const float *bias, *g, *mu, *root, *sigma; };

static inline GmmP leaf(void* const* d_in, int base) {
    GmmP p;
    p.bias  = (const float*)d_in[base + 0];
    p.g     = (const float*)d_in[base + 1];
    p.mu    = (const float*)d_in[base + 2];
    p.root  = (const float*)d_in[base + 3];
    p.sigma = (const float*)d_in[base + 4];
    return p;
}

static inline int cdiv(long long a, int b) { return (int)((a + b - 1) / b); }

static void run_conv(hipStream_t st, const __bf16* xbf, const int* edges, const float* ea,
                     int N, int E, int cin, int M, const GmmP& p, float* xout,
                     __bf16* Wt, float* Y, float* ew, float* agg, float* cnt) {
    int cols = 4 * M;
    int wtot = cols * cin;
    prep_weights<<<cdiv(wtot, 256), 256, 0, st>>>(p.g, p.root, Wt, cin, M);

    int tiles = ((N + 15) / 16) * (cols / 16);
    gemm_xw_bf16<<<cdiv(tiles, 8), 256, 0, st>>>(xbf, Wt, Y, N, cin, cols, tiles);

    edge_weights<<<cdiv(E, 256), 256, 0, st>>>(ea, p.mu, p.sigma, ew, E);

    size_t aggN = (size_t)N * M;
    zero_f32<<<cdiv((long long)aggN, 256), 256, 0, st>>>(agg, aggN);
    zero_f32<<<cdiv(N, 256), 256, 0, st>>>(cnt, (size_t)N);

    scatter_msgs<<<cdiv((long long)E * M, 256), 256, 0, st>>>(edges, Y, ew, agg, E, M, cols);
    count_edges<<<cdiv(E, 256), 256, 0, st>>>(edges, cnt, E);

    finalize_k<<<cdiv((long long)N * M, 256), 256, 0, st>>>(agg, cnt, Y, p.bias, xout, N, M, cols);
}

extern "C" void kernel_launch(void* const* d_in, const int* in_sizes, int n_in,
                              void* d_out, int out_size, void* d_ws, size_t ws_size,
                              hipStream_t stream) {
    (void)in_sizes; (void)n_in; (void)out_size; (void)ws_size;

    // ---- inputs (setup_inputs insertion order) ----
    const float* x_in  = (const float*)d_in[0];   // [N7, 32]
    const int* edges7  = (const int*)d_in[1];     // [2, E7]
    const float* ea7   = (const float*)d_in[2];   // [E7, 2]
    const int* edges6  = (const int*)d_in[3];
    const float* ea6   = (const float*)d_in[4];
    const int* edges5  = (const int*)d_in[5];
    const float* ea5   = (const float*)d_in[6];
    const int* neigh6  = (const int*)d_in[7];     // [N6, 7]
    const int* neigh5  = (const int*)d_in[8];     // [N5, 7]
    const int* ups6    = (const int*)d_in[9];     // [N6-N5, 2]
    const int* ups7    = (const int*)d_in[10];    // [N7-N6, 2]

    // params pytree leaves (dicts sorted by key): dec, enc, fc; each conv as
    // bias, g, mu, root, sigma.
    GmmP dec00 = leaf(d_in, 11), dec01 = leaf(d_in, 16);
    GmmP dec10 = leaf(d_in, 21), dec11 = leaf(d_in, 26);
    GmmP enc00 = leaf(d_in, 31), enc01 = leaf(d_in, 36);
    GmmP enc10 = leaf(d_in, 41), enc11 = leaf(d_in, 46);
    GmmP enc20 = leaf(d_in, 51), enc21 = leaf(d_in, 56);
    const float* fc_b = (const float*)d_in[61];   // [2]
    const float* fc_w = (const float*)d_in[62];   // [32, 2]

    // ---- workspace carve-up (deterministic, 256B aligned) ----
    char* ws = (char*)d_ws;
    size_t off = 0;
    auto take = [&](size_t bytes) -> char* {
        char* p = ws + off;
        off += (bytes + 255) & ~(size_t)255;
        return p;
    };
    __bf16* XBF = (__bf16*)take((size_t)CN7 * 96 * 2);   // bf16 conv input (max N7 x 96)
    __bf16* WT  = (__bf16*)take((size_t)128 * 256 * 2);  // transposed combined weights
    float*  Y   = (float*) take((size_t)CN7 * 128 * 4);  // GEMM output [N, 4M]
    float*  EW  = (float*) take((size_t)CE7 * 3 * 4);    // edge Gaussian weights
    float*  AGG = (float*) take((size_t)CN7 * 32 * 4);   // segment-sum accumulator
    float*  CNT = (float*) take((size_t)CN7 * 4);        // per-dst edge counts
    float*  X1  = (float*) take((size_t)CN7 * 32 * 4);   // f32 conv output
    float*  SK7 = (float*) take((size_t)CN7 * 32 * 4);   // skip (level 7)
    float*  SK6 = (float*) take((size_t)CN6 * 64 * 4);   // skip (level 6)

    // ---- encoder, level 7 (32 -> 32 -> 32) ----
    f32_to_bf16<<<cdiv((long long)CN7 * 32, 256), 256, 0, stream>>>(x_in, XBF, (size_t)CN7 * 32);
    run_conv(stream, XBF, edges7, ea7, CN7, CE7, 32, 32, enc00, X1, WT, Y, EW, AGG, CNT);
    f32_to_bf16<<<cdiv((long long)CN7 * 32, 256), 256, 0, stream>>>(X1, XBF, (size_t)CN7 * 32);
    run_conv(stream, XBF, edges7, ea7, CN7, CE7, 32, 32, enc01, X1, WT, Y, EW, AGG, CNT);
    hipMemcpyAsync(SK7, X1, (size_t)CN7 * 32 * 4, hipMemcpyDeviceToDevice, stream);
    hex_pool<<<cdiv((long long)CN6 * 32, 256), 256, 0, stream>>>(X1, neigh6, XBF, CN6, 32);

    // ---- encoder, level 6 (32 -> 64 -> 64) ----
    run_conv(stream, XBF, edges6, ea6, CN6, CE6, 32, 64, enc10, X1, WT, Y, EW, AGG, CNT);
    f32_to_bf16<<<cdiv((long long)CN6 * 64, 256), 256, 0, stream>>>(X1, XBF, (size_t)CN6 * 64);
    run_conv(stream, XBF, edges6, ea6, CN6, CE6, 64, 64, enc11, X1, WT, Y, EW, AGG, CNT);
    hipMemcpyAsync(SK6, X1, (size_t)CN6 * 64 * 4, hipMemcpyDeviceToDevice, stream);
    hex_pool<<<cdiv((long long)CN5 * 64, 256), 256, 0, stream>>>(X1, neigh5, XBF, CN5, 64);

    // ---- encoder, level 5 (64 -> 64 -> 64) ----
    run_conv(stream, XBF, edges5, ea5, CN5, CE5, 64, 64, enc20, X1, WT, Y, EW, AGG, CNT);
    f32_to_bf16<<<cdiv((long long)CN5 * 64, 256), 256, 0, stream>>>(X1, XBF, (size_t)CN5 * 64);
    run_conv(stream, XBF, edges5, ea5, CN5, CE5, 64, 64, enc21, X1, WT, Y, EW, AGG, CNT);

    // ---- decoder, level 6: unpool + skip -> [N6, 128] -> 64 -> 64 ----
    hex_unpool<<<cdiv((long long)CN6 * 64, 256), 256, 0, stream>>>(X1, ups6, XBF, CN5, CN6, 64, 128);
    concat_skip<<<cdiv((long long)CN6 * 64, 256), 256, 0, stream>>>(SK6, XBF, CN6, 64, 128, 64);
    run_conv(stream, XBF, edges6, ea6, CN6, CE6, 128, 64, dec00, X1, WT, Y, EW, AGG, CNT);
    f32_to_bf16<<<cdiv((long long)CN6 * 64, 256), 256, 0, stream>>>(X1, XBF, (size_t)CN6 * 64);
    run_conv(stream, XBF, edges6, ea6, CN6, CE6, 64, 64, dec01, X1, WT, Y, EW, AGG, CNT);

    // ---- decoder, level 7: unpool + skip -> [N7, 96] -> 32 -> 32 ----
    hex_unpool<<<cdiv((long long)CN7 * 64, 256), 256, 0, stream>>>(X1, ups7, XBF, CN6, CN7, 64, 96);
    concat_skip<<<cdiv((long long)CN7 * 32, 256), 256, 0, stream>>>(SK7, XBF, CN7, 32, 96, 64);
    run_conv(stream, XBF, edges7, ea7, CN7, CE7, 96, 32, dec10, X1, WT, Y, EW, AGG, CNT);
    f32_to_bf16<<<cdiv((long long)CN7 * 32, 256), 256, 0, stream>>>(X1, XBF, (size_t)CN7 * 32);
    run_conv(stream, XBF, edges7, ea7, CN7, CE7, 32, 32, dec11, X1, WT, Y, EW, AGG, CNT);

    // ---- final FC + log_softmax ----
    fc_logsoftmax<<<cdiv(CN7, 256), 256, 0, stream>>>(X1, fc_w, fc_b, (float*)d_out, CN7);
}
